// SpecializedGrid3x3_64476049047714
// MI455X (gfx1250) — compile-verified, matrix operand order checked by reading, not run on hardware
//
#include <hip/hip_runtime.h>
#include <hip/hip_bf16.h>

typedef __bf16 bf16;
typedef __attribute__((ext_vector_type(16))) __bf16 v16bf;
typedef __attribute__((ext_vector_type(8)))  float  v8f;

#define D_  1024
#define B_  1024
#define FS_ 2048
#define FE_ 8192

#define TM 128
#define TN 128
#define TK 32
#define LDSTR 40   // halves per LDS row (32 + 8 pad, keeps 16B alignment)

static __device__ inline bf16 f2bf(float f) {
    union { float f; unsigned int u; } a; a.f = f;
    unsigned int r = a.u + 0x7FFFu + ((a.u >> 16) & 1u);  // round-to-nearest-even
    union { unsigned short u; bf16 b; } c; c.u = (unsigned short)(r >> 16);
    return c.b;
}

union Frag { v16bf v; uint4 q[2]; };

// CDNA5 async global->LDS copy (bypasses VGPRs, tracked by ASYNCcnt).
// lds_off: per-lane LDS byte address (hardware adds LDS_BASE); gaddr: 64-bit global addr.
static __device__ inline void async_ld_b128(unsigned lds_off, const void* gptr) {
    unsigned long long ga = (unsigned long long)(uintptr_t)gptr;
    asm volatile("global_load_async_to_lds_b128 %0, %1, off"
                 :: "v"(lds_off), "v"(ga) : "memory");
}

struct GemmCfg {
    int M, N, K;
    int node_mul, node_add;   // grid-node id = z*node_mul + node_add
    int a_by_slot;            // A indexed by z (1) or by gnode (0)
    int out_by_slot;          // bf16 out indexed by z (1) or gnode (0)
    int relu;
    int write_bf16;
    int write_f32;
    int add_resid;            // += resid[gnode][m][n] (f32)
    int mem_update;           // memory-node EMA blend on nodes {0,2,6,8}
};

// ---------------------------------------------------------------------------
// bf16 WMMA GEMM:  out[z/gnode] = epilogue( A[.] (MxK) * W[z] (NxK)^T + bias[z] )
// block tile 128x128, 8 waves (4 in M x 2 in N), wave tile 32x64 = 2x4 WMMA tiles
// Double-buffered LDS filled by global_load_async_to_lds_b128.
// ---------------------------------------------------------------------------
__global__ __launch_bounds__(256, 2)
void gemm_bf16_wmma(const bf16* __restrict__ Ab, const bf16* __restrict__ Wb,
                    const float* __restrict__ bias, const float* __restrict__ resid,
                    float* __restrict__ outF, bf16* __restrict__ outB,
                    float* __restrict__ mem, GemmCfg c)
{
    __shared__ __align__(16) bf16 lA[2][TM * LDSTR];
    __shared__ __align__(16) bf16 lW[2][TN * LDSTR];

    const int tid  = threadIdx.x;
    const int wave = tid >> 5, lane = tid & 31;
    const int h2 = lane >> 4, l16 = lane & 15;
    const int wm = wave & 3, wn = wave >> 2;          // 4 x 2 wave grid
    const int z = blockIdx.z;
    const int gnode = z * c.node_mul + c.node_add;

    const bf16* Ag = Ab + (size_t)(c.a_by_slot ? z : gnode) * c.M * c.K;
    const bf16* Wg = Wb + (size_t)z * c.N * c.K;

    const int m0 = blockIdx.y * TM;
    const int n0 = blockIdx.x * TN;

    v8f acc[2][4];
    const v8f vzero = {0.f, 0.f, 0.f, 0.f, 0.f, 0.f, 0.f, 0.f};
#pragma unroll
    for (int i = 0; i < 2; i++)
#pragma unroll
        for (int j = 0; j < 4; j++) acc[i][j] = vzero;

    // cooperative tile load plan: 512 x 16B chunks per tile, 2 per thread
    const int c0 = tid * 2, c1 = tid * 2 + 1;
    const int r0 = c0 >> 2, s0 = c0 & 3;
    const int r1 = c1 >> 2, s1 = c1 & 3;

    // per-wave: exactly 4 async-load instructions per tile (A x2 + W x2)
    auto issue_tile = [&](int k0, int buf) {
        async_ld_b128((unsigned)(uintptr_t)&lA[buf][r0 * LDSTR + s0 * 8],
                      &Ag[(size_t)(m0 + r0) * c.K + k0 + s0 * 8]);
        async_ld_b128((unsigned)(uintptr_t)&lA[buf][r1 * LDSTR + s1 * 8],
                      &Ag[(size_t)(m0 + r1) * c.K + k0 + s1 * 8]);
        async_ld_b128((unsigned)(uintptr_t)&lW[buf][r0 * LDSTR + s0 * 8],
                      &Wg[(size_t)(n0 + r0) * c.K + k0 + s0 * 8]);
        async_ld_b128((unsigned)(uintptr_t)&lW[buf][r1 * LDSTR + s1 * 8],
                      &Wg[(size_t)(n0 + r1) * c.K + k0 + s1 * 8]);
    };

    const int nk = c.K / TK;
    issue_tile(0, 0);

    for (int t = 0; t < nk; t++) {
        const int buf = t & 1;
        if (t + 1 < nk) {
            issue_tile((t + 1) * TK, (t + 1) & 1);   // overlap next copy with compute
            asm volatile("s_wait_asynccnt 0x4" ::: "memory");  // tile t landed
            if (t + 2 < nk) {                        // warm L2 for tile t+2
                __builtin_prefetch(&Ag[(size_t)(m0 + r0) * c.K + (t + 2) * TK + s0 * 8], 0, 3);
                __builtin_prefetch(&Wg[(size_t)(n0 + r0) * c.K + (t + 2) * TK + s0 * 8], 0, 3);
            }
        } else {
            asm volatile("s_wait_asynccnt 0x0" ::: "memory");
        }
        __syncthreads();

        Frag fa[2], fb[4];
#pragma unroll
        for (int ms = 0; ms < 2; ms++) {
            const bf16* lr = &lA[buf][(wm * 32 + ms * 16 + l16) * LDSTR];
            fa[ms].q[0] = *(const uint4*)(lr + h2 * 8);        // K = 0..7   (+8 for hi half)
            fa[ms].q[1] = *(const uint4*)(lr + 16 + h2 * 8);   // K = 16..23 (+8 for hi half)
        }
#pragma unroll
        for (int ns = 0; ns < 4; ns++) {
            const bf16* lr = &lW[buf][(wn * 64 + ns * 16 + l16) * LDSTR];
            fb[ns].q[0] = *(const uint4*)(lr + h2 * 8);
            fb[ns].q[1] = *(const uint4*)(lr + 16 + h2 * 8);
        }
#pragma unroll
        for (int ms = 0; ms < 2; ms++)
#pragma unroll
            for (int ns = 0; ns < 4; ns++)
                acc[ms][ns] = __builtin_amdgcn_wmma_f32_16x16x32_bf16(
                    false, fa[ms].v, false, fb[ns].v, (short)0, acc[ms][ns], false, false);

        __syncthreads();   // buffer `buf` free for tile t+2's async writes
    }

    // ---- epilogue ----
    const int is_mem = (c.mem_update && gnode != 4 && (gnode & 1) == 0);
    int slot = gnode >> 1; if (gnode > 4) slot--;         // 0,2,6,8 -> 0,1,2,3

    const size_t outNodeF  = (size_t)gnode * c.M * c.N;
    const size_t outNodeB  = (size_t)(c.out_by_slot ? z : gnode) * c.M * c.N;
    const size_t residNode = (size_t)gnode * c.M * c.N;
    const float* biasRow   = bias + (size_t)z * c.N;

#pragma unroll
    for (int ms = 0; ms < 2; ms++) {
#pragma unroll
        for (int ns = 0; ns < 4; ns++) {
            const int col = n0 + wn * 64 + ns * 16 + l16;
            const float bv = biasRow[col];
#pragma unroll
            for (int r = 0; r < 8; r++) {
                const int row = m0 + wm * 32 + ms * 16 + h2 * 8 + r;
                float vv = acc[ms][ns][r] + bv;
                const size_t idx = (size_t)row * c.N + col;
                if (c.add_resid) vv += resid[residNode + idx];
                if (is_mem) {
                    const size_t mi = (size_t)slot * c.M * c.N + idx;
                    const float nm = 0.7f * mem[mi] + 0.3f * vv;
                    mem[mi] = nm;
                    vv = nm;
                }
                if (c.relu) vv = vv > 0.f ? vv : 0.f;
                if (c.write_f32)  outF[outNodeF + idx] = vv;
                if (c.write_bf16) outB[outNodeB + idx] = f2bf(vv);
            }
        }
    }
}

// ---------------------------------------------------------------------------
// adjacency sum over static 3x3 grid + LayerNorm; writes integ (f32) + xn (bf16)
// ---------------------------------------------------------------------------
__global__ void integ_ln_kernel(const float* __restrict__ S,
                                const float* __restrict__ sc, const float* __restrict__ bi,
                                float* __restrict__ integ, bf16* __restrict__ xn)
{
    const int b = blockIdx.x, node = blockIdx.y, tid = threadIdx.x;
    const int r = node / 3, cc = node % 3;
    int nbr[4]; int nn = 0;
    if (r > 0)  nbr[nn++] = (r - 1) * 3 + cc;
    if (r < 2)  nbr[nn++] = (r + 1) * 3 + cc;
    if (cc > 0) nbr[nn++] = r * 3 + cc - 1;
    if (cc < 2) nbr[nn++] = r * 3 + cc + 1;

    float v[4], s1 = 0.f, s2 = 0.f;
#pragma unroll
    for (int j = 0; j < 4; j++) {
        const int d = tid + j * 256;
        float a = 0.f;
        for (int q = 0; q < nn; q++)
            a += S[((size_t)nbr[q] * B_ + b) * D_ + d];
        v[j] = a; s1 += a; s2 += a * a;
    }
    __shared__ float red[256];
    red[tid] = s1; __syncthreads();
    for (int o = 128; o > 0; o >>= 1) { if (tid < o) red[tid] += red[tid + o]; __syncthreads(); }
    const float mu = red[0] * (1.f / D_); __syncthreads();
    red[tid] = s2; __syncthreads();
    for (int o = 128; o > 0; o >>= 1) { if (tid < o) red[tid] += red[tid + o]; __syncthreads(); }
    const float var = red[0] * (1.f / D_) - mu * mu;
    const float rstd = rsqrtf(var + 1e-5f);
#pragma unroll
    for (int j = 0; j < 4; j++) {
        const int d = tid + j * 256;
        const size_t o = ((size_t)node * B_ + b) * D_ + d;
        integ[o] = v[j];
        xn[o] = f2bf((v[j] - mu) * rstd * sc[node * D_ + d] + bi[node * D_ + d]);
    }
}

// plain per-node LayerNorm: x (f32) -> xn2 (bf16)
__global__ void ln_kernel(const float* __restrict__ X,
                          const float* __restrict__ sc, const float* __restrict__ bi,
                          bf16* __restrict__ out)
{
    const int b = blockIdx.x, node = blockIdx.y, tid = threadIdx.x;
    float v[4], s1 = 0.f, s2 = 0.f;
#pragma unroll
    for (int j = 0; j < 4; j++) {
        const int d = tid + j * 256;
        const float a = X[((size_t)node * B_ + b) * D_ + d];
        v[j] = a; s1 += a; s2 += a * a;
    }
    __shared__ float red[256];
    red[tid] = s1; __syncthreads();
    for (int o = 128; o > 0; o >>= 1) { if (tid < o) red[tid] += red[tid + o]; __syncthreads(); }
    const float mu = red[0] * (1.f / D_); __syncthreads();
    red[tid] = s2; __syncthreads();
    for (int o = 128; o > 0; o >>= 1) { if (tid < o) red[tid] += red[tid + o]; __syncthreads(); }
    const float var = red[0] * (1.f / D_) - mu * mu;
    const float rstd = rsqrtf(var + 1e-5f);
#pragma unroll
    for (int j = 0; j < 4; j++) {
        const int d = tid + j * 256;
        out[((size_t)node * B_ + b) * D_ + d] =
            f2bf((v[j] - mu) * rstd * sc[node * D_ + d] + bi[node * D_ + d]);
    }
}

__global__ void init_s_kernel(const float* __restrict__ seed, float* __restrict__ S)
{
    const size_t i = (size_t)blockIdx.x * 256 + threadIdx.x;
    const size_t total = (size_t)9 * B_ * D_;
    if (i >= total) return;
    const size_t node = i / ((size_t)B_ * D_);
    S[i] = (node == 4) ? seed[i - (size_t)4 * B_ * D_] : 0.f;
}

__global__ void cvt_f32_bf16(const float* __restrict__ in, bf16* __restrict__ out, long n)
{
    const long i = ((long)blockIdx.x * 256 + threadIdx.x) * 4;
    if (i + 3 < n) {
        const float4 f = *(const float4*)(in + i);
        out[i]     = f2bf(f.x); out[i + 1] = f2bf(f.y);
        out[i + 2] = f2bf(f.z); out[i + 3] = f2bf(f.w);
    } else {
        for (long j = i; j < n; j++) out[j] = f2bf(in[j]);
    }
}

// ---------------------------------------------------------------------------
extern "C" void kernel_launch(void* const* d_in, const int* in_sizes, int n_in,
                              void* d_out, int out_size, void* d_ws, size_t ws_size,
                              hipStream_t stream)
{
    const float* seed  = (const float*)d_in[0];
    const float* ln1_s = (const float*)d_in[1];
    const float* ln1_b = (const float*)d_in[2];
    const float* v_w   = (const float*)d_in[3];
    const float* v_b   = (const float*)d_in[4];
    const float* o_w   = (const float*)d_in[5];
    const float* o_b   = (const float*)d_in[6];
    const float* ln2_s = (const float*)d_in[7];
    const float* ln2_b = (const float*)d_in[8];
    const float* w1_s  = (const float*)d_in[9];
    const float* b1_s  = (const float*)d_in[10];
    const float* w2_s  = (const float*)d_in[11];
    const float* b2_s  = (const float*)d_in[12];
    const float* w1_e  = (const float*)d_in[13];
    const float* b1_e  = (const float*)d_in[14];
    const float* w2_e  = (const float*)d_in[15];
    const float* b2_e  = (const float*)d_in[16];
    const int STEPS = 3;   // matches reference setup_inputs()

    float* S = (float*)d_out;   // S lives in d_out: [9, B, 1, D]

    // workspace layout
    char* ws = (char*)d_ws;
    size_t off = 0;
    auto take = [&](size_t bytes) { char* p = ws + off; off = (off + bytes + 255) & ~(size_t)255; return p; };
    const size_t nVW  = (size_t)9 * D_ * D_;
    const size_t nW1S = (size_t)5 * FS_ * D_;
    const size_t nW1E = (size_t)4 * FE_ * D_;
    bf16* v_wb  = (bf16*)take(nVW  * 2);
    bf16* o_wb  = (bf16*)take(nVW  * 2);
    bf16* w1_sb = (bf16*)take(nW1S * 2);
    bf16* w2_sb = (bf16*)take(nW1S * 2);
    bf16* w1_eb = (bf16*)take(nW1E * 2);
    bf16* w2_eb = (bf16*)take(nW1E * 2);
    bf16* xn    = (bf16*)take((size_t)9 * B_ * D_  * 2);
    bf16* vbuf  = (bf16*)take((size_t)9 * B_ * D_  * 2);
    bf16* xn2   = (bf16*)take((size_t)9 * B_ * D_  * 2);
    bf16* h_s   = (bf16*)take((size_t)5 * B_ * FS_ * 2);
    bf16* h_e   = (bf16*)take((size_t)4 * B_ * FE_ * 2);
    float* integ = (float*)take((size_t)9 * B_ * D_ * 4);
    float* xbuf  = (float*)take((size_t)9 * B_ * D_ * 4);
    float* mem   = (float*)take((size_t)4 * B_ * D_ * 4);

    // one-time per call: bf16 weight conversion + state init
    auto cvt = [&](const float* in, bf16* out, size_t n) {
        cvt_f32_bf16<<<dim3((unsigned)((n / 4 + 255) / 256)), 256, 0, stream>>>(in, out, (long)n);
    };
    cvt(v_w,  v_wb,  nVW);
    cvt(o_w,  o_wb,  nVW);
    cvt(w1_s, w1_sb, nW1S);
    cvt(w2_s, w2_sb, nW1S);
    cvt(w1_e, w1_eb, nW1E);
    cvt(w2_e, w2_eb, nW1E);
    hipMemsetAsync(mem, 0, (size_t)4 * B_ * D_ * 4, stream);
    {
        const size_t total = (size_t)9 * B_ * D_;
        init_s_kernel<<<dim3((unsigned)((total + 255) / 256)), 256, 0, stream>>>(seed, S);
    }

    auto gemm = [&](int nodes, const bf16* A, const bf16* W, const float* bias,
                    const float* resid, float* oF, bf16* oB, float* memp, GemmCfg c) {
        dim3 grid(c.N / TN, c.M / TM, nodes);
        gemm_bf16_wmma<<<grid, 256, 0, stream>>>(A, W, bias, resid, oF, oB, memp, c);
    };

    for (int s = 0; s < STEPS; s++) {
        // integ = A_adj @ S ; xn = LN1(integ)
        integ_ln_kernel<<<dim3(B_, 9), 256, 0, stream>>>(S, ln1_s, ln1_b, integ, xn);

        // v = xn @ v_w^T + v_b           (bf16 out, by gnode)
        GemmCfg cv{B_, D_, D_, 1, 0, 0, 0, 0, 1, 0, 0, 0};
        gemm(9, xn, v_wb, v_b, nullptr, nullptr, vbuf, nullptr, cv);

        // x = integ + (v @ o_w^T + o_b); memory-node EMA blend; f32 out
        GemmCfg co{B_, D_, D_, 1, 0, 0, 0, 0, 0, 1, 1, 1};
        gemm(9, vbuf, o_wb, o_b, integ, xbuf, nullptr, mem, co);

        // xn2 = LN2(x)
        ln_kernel<<<dim3(B_, 9), 256, 0, stream>>>(xbuf, ln2_s, ln2_b, xn2);

        // h = relu(xn2 @ w1^T + b1)
        GemmCfg c1s{B_, FS_, D_, 2, 0, 0, 1, 1, 1, 0, 0, 0};
        gemm(5, xn2, w1_sb, b1_s, nullptr, nullptr, h_s, nullptr, c1s);
        GemmCfg c1e{B_, FE_, D_, 2, 1, 0, 1, 1, 1, 0, 0, 0};
        gemm(4, xn2, w1_eb, b1_e, nullptr, nullptr, h_e, nullptr, c1e);

        // S = x + (h @ w2^T + b2)
        GemmCfg c2s{B_, D_, FS_, 2, 0, 1, 0, 0, 0, 1, 1, 0};
        gemm(5, h_s, w2_sb, b2_s, xbuf, S, nullptr, nullptr, c2s);
        GemmCfg c2e{B_, D_, FE_, 2, 1, 1, 0, 0, 0, 1, 1, 0};
        gemm(4, h_e, w2_eb, b2_e, xbuf, S, nullptr, nullptr, c2e);
    }
}
